// BiDirectionalAttention_3418793967993
// MI455X (gfx1250) — matrix operand order repaired and checked by reading, not compiled, not run on hardware
//
#include <hip/hip_runtime.h>
#include <stdint.h>

// ---------------------------------------------------------------------------
// Types for CDNA5 WMMA (wave32, v_wmma_f32_16x16x32_bf16)
// ---------------------------------------------------------------------------
typedef __bf16 bf16_t;
typedef bf16_t v16bf __attribute__((ext_vector_type(16)));
typedef float  v8f   __attribute__((ext_vector_type(8)));
typedef unsigned int u32x4 __attribute__((ext_vector_type(4)));
typedef unsigned int u32x2 __attribute__((ext_vector_type(2)));
typedef float        f32x4 __attribute__((ext_vector_type(4)));

union FragBF { u32x4 q[2]; v16bf v; };

__device__ __forceinline__ unsigned short f2bf(float f) {
  // round-to-nearest-even f32 -> bf16
  unsigned u = __builtin_bit_cast(unsigned, f);
  unsigned r = u + 0x7fffu + ((u >> 16) & 1u);
  return (unsigned short)(r >> 16);
}

constexpr int CL = 16384;   // c_len
constexpr int QL = 1024;    // q_len
constexpr int DD = 1024;    // d
constexpr float LOG2E = 1.4426950408889634f;

// ---------------------------------------------------------------------------
// prep_u: Ub[q][d] = bf16(U), UbT[d][q] = bf16(U), uq[q] = U.w_q + sum(bias)
// grid: QL blocks x 256 threads (4 d-elements / thread)
// ---------------------------------------------------------------------------
__global__ void __launch_bounds__(256) prep_u_kernel(
    const float* __restrict__ U, const float* __restrict__ w_q,
    const float* __restrict__ b_q, const float* __restrict__ b_c,
    const float* __restrict__ b_qc,
    unsigned short* __restrict__ Ub, unsigned short* __restrict__ UbT,
    float* __restrict__ uq) {
  const int q = blockIdx.x;
  const int t = threadIdx.x;
  const int d0 = t * 4;
  f32x4 u4 = *(const f32x4*)(U + (size_t)q * DD + d0);
  f32x4 w4 = *(const f32x4*)(w_q + d0);
  unsigned short h0 = f2bf(u4.x), h1 = f2bf(u4.y), h2 = f2bf(u4.z), h3 = f2bf(u4.w);
  u32x2 pk;
  pk.x = (unsigned)h0 | ((unsigned)h1 << 16);
  pk.y = (unsigned)h2 | ((unsigned)h3 << 16);
  *(u32x2*)(Ub + (size_t)q * DD + d0) = pk;
  UbT[(size_t)(d0 + 0) * QL + q] = h0;
  UbT[(size_t)(d0 + 1) * QL + q] = h1;
  UbT[(size_t)(d0 + 2) * QL + q] = h2;
  UbT[(size_t)(d0 + 3) * QL + q] = h3;
  float dot = u4.x * w4.x + u4.y * w4.y + u4.z * w4.z + u4.w * w4.w;
  for (int m = 16; m; m >>= 1) dot += __shfl_xor(dot, m, 32);
  __shared__ float red[8];
  if ((t & 31) == 0) red[t >> 5] = dot;
  __syncthreads();
  if (t == 0) {
    float s = 0.f;
    for (int i = 0; i < 8; ++i) s += red[i];
    uq[q] = s + b_q[0] + b_c[0] + b_qc[0];
  }
}

// ---------------------------------------------------------------------------
// prep_h: Hs[c][d] = bf16(H * w_qc), hc[c] = H.w_c
// grid: CL blocks x 256 threads
// ---------------------------------------------------------------------------
__global__ void __launch_bounds__(256) prep_h_kernel(
    const float* __restrict__ H, const float* __restrict__ w_qc,
    const float* __restrict__ w_c,
    unsigned short* __restrict__ Hs, float* __restrict__ hc) {
  const int c = blockIdx.x;
  const int t = threadIdx.x;
  const int d0 = t * 4;
  f32x4 h4  = *(const f32x4*)(H + (size_t)c * DD + d0);
  f32x4 wqc = *(const f32x4*)(w_qc + d0);
  f32x4 wc  = *(const f32x4*)(w_c + d0);
  unsigned short h0 = f2bf(h4.x * wqc.x), h1 = f2bf(h4.y * wqc.y);
  unsigned short h2 = f2bf(h4.z * wqc.z), h3 = f2bf(h4.w * wqc.w);
  u32x2 pk;
  pk.x = (unsigned)h0 | ((unsigned)h1 << 16);
  pk.y = (unsigned)h2 | ((unsigned)h3 << 16);
  *(u32x2*)(Hs + (size_t)c * DD + d0) = pk;
  float dot = h4.x * wc.x + h4.y * wc.y + h4.z * wc.z + h4.w * wc.w;
  for (int m = 16; m; m >>= 1) dot += __shfl_xor(dot, m, 32);
  __shared__ float red[8];
  if ((t & 31) == 0) red[t >> 5] = dot;
  __syncthreads();
  if (t == 0) {
    float s = 0.f;
    for (int i = 0; i < 8; ++i) s += red[i];
    hc[c] = s;
  }
}

// ---------------------------------------------------------------------------
// attn: fused  S = Hs @ Ub^T (+hc+uq)  -> softmax over q  -> P @ U  -> outU
// also writes bmax[c] = max_q S.
// Workgroup: 256 threads = 8 waves, handles 32 c-rows x 1024 q (x 1024 d out).
// Wave (rt = w&1, g = w>>1 grouping): rt selects 16-row tile, g selects 256
// columns; each wave owns 16 WMMA accumulator tiles.
// grid: CL/32 = 512 blocks
// ---------------------------------------------------------------------------
__global__ void __launch_bounds__(256) attn_kernel(
    const unsigned short* __restrict__ Hs, const unsigned short* __restrict__ Ub,
    const unsigned short* __restrict__ UbT, const float* __restrict__ uq,
    const float* __restrict__ hc, float* __restrict__ outU,
    float* __restrict__ bmax) {
  __shared__ unsigned short Plds[32 * QL];   // 64 KB bf16 P tile
  __shared__ float pmax[4][32];
  __shared__ float psum[4][32];

  const int tid  = threadIdx.x;
  const int wave = tid >> 5;
  const int lane = tid & 31;
  const int rt   = wave & 1;        // row tile (16 rows)
  const int g    = wave >> 1;       // column group (256 cols)
  const int hi   = lane >> 4;       // which 16-lane half
  const int ln   = lane & 15;
  const int cbase = blockIdx.x * 32;

  // ---------------- GEMM1: S = Hs @ Ub^T ----------------
  v8f C[16];
  v8f vzero = {};
#pragma unroll
  for (int t = 0; t < 16; ++t) C[t] = vzero;

  const unsigned short* Arow = Hs + (size_t)(cbase + rt * 16 + ln) * DD;

  for (int kb = 0; kb < 32; ++kb) {
    const int k0 = kb * 32;
    if (kb < 31) __builtin_prefetch(Arow + k0 + 32, 0, 1);
    FragBF A;
    // 16-bit A 16x32 layout: lane holds row (l&15); hi half owns K+8 offset
    A.q[0] = *(const u32x4*)(Arow + k0 + 8 * hi);
    A.q[1] = *(const u32x4*)(Arow + k0 + 16 + 8 * hi);
#pragma unroll
    for (int t = 0; t < 16; ++t) {
      const int q = g * 256 + t * 16 + ln;           // B column (q index)
      const unsigned short* bp = Ub + (size_t)q * DD + k0 + 8 * hi;
      FragBF B;                                      // lanes0-15: K0..7,16..23, lanes16-31: K8..15,24..31
      B.q[0] = *(const u32x4*)(bp);
      B.q[1] = *(const u32x4*)(bp + 16);
      C[t] = __builtin_amdgcn_wmma_f32_16x16x32_bf16(
          false, A.v, false, B.v, (short)0, C[t], false, false);
    }
  }

  // ---------------- epilogue: bias add ----------------
  float hcv[8];
#pragma unroll
  for (int r = 0; r < 8; ++r) hcv[r] = hc[cbase + rt * 16 + 8 * hi + r];
#pragma unroll
  for (int t = 0; t < 16; ++t) {
    float uqv = uq[g * 256 + t * 16 + ln];
#pragma unroll
    for (int r = 0; r < 8; ++r) C[t][r] += hcv[r] + uqv;
  }

  // ---------------- row max (softmax over q) ----------------
  float mx[8];
#pragma unroll
  for (int r = 0; r < 8; ++r) mx[r] = -3.0e38f;
#pragma unroll
  for (int t = 0; t < 16; ++t)
#pragma unroll
    for (int r = 0; r < 8; ++r) mx[r] = fmaxf(mx[r], C[t][r]);
#pragma unroll
  for (int r = 0; r < 8; ++r)
    for (int m = 1; m < 16; m <<= 1) mx[r] = fmaxf(mx[r], __shfl_xor(mx[r], m, 32));
  if (ln == 0) {
#pragma unroll
    for (int r = 0; r < 8; ++r) pmax[g][rt * 16 + 8 * hi + r] = mx[r];
  }
  __syncthreads();
  float fm[8];
#pragma unroll
  for (int r = 0; r < 8; ++r) {
    int m = rt * 16 + 8 * hi + r;
    fm[r] = fmaxf(fmaxf(pmax[0][m], pmax[1][m]), fmaxf(pmax[2][m], pmax[3][m]));
  }
  if (g == 0 && ln == 0) {
#pragma unroll
    for (int r = 0; r < 8; ++r) bmax[cbase + rt * 16 + 8 * hi + r] = fm[r];
  }

  // ---------------- exp + row sum ----------------
  float sm[8];
#pragma unroll
  for (int r = 0; r < 8; ++r) sm[r] = 0.f;
#pragma unroll
  for (int t = 0; t < 16; ++t)
#pragma unroll
    for (int r = 0; r < 8; ++r) {
      float e = exp2f((C[t][r] - fm[r]) * LOG2E);
      C[t][r] = e;
      sm[r] += e;
    }
#pragma unroll
  for (int r = 0; r < 8; ++r)
    for (int m = 1; m < 16; m <<= 1) sm[r] += __shfl_xor(sm[r], m, 32);
  if (ln == 0) {
#pragma unroll
    for (int r = 0; r < 8; ++r) psum[g][rt * 16 + 8 * hi + r] = sm[r];
  }
  __syncthreads();
  float inv[8];
#pragma unroll
  for (int r = 0; r < 8; ++r) {
    int m = rt * 16 + 8 * hi + r;
    inv[r] = 1.0f / (psum[0][m] + psum[1][m] + psum[2][m] + psum[3][m]);
  }

  // ---------------- P (bf16) -> LDS ----------------
#pragma unroll
  for (int t = 0; t < 16; ++t) {
    const int qc = g * 256 + t * 16 + ln;
#pragma unroll
    for (int r = 0; r < 8; ++r) {
      const int m = rt * 16 + 8 * hi + r;
      Plds[m * QL + qc] = f2bf(C[t][r] * inv[r]);
    }
  }
  __syncthreads();

  // ---------------- GEMM2: outU = P @ U  (B from UbT, contiguous K=q) -----
  v8f O[16];
#pragma unroll
  for (int t = 0; t < 16; ++t) O[t] = vzero;

  const unsigned short* Ald = Plds + (rt * 16 + ln) * QL;
  for (int kb = 0; kb < 32; ++kb) {
    const int k0 = kb * 32;
    FragBF A;
    A.q[0] = *(const u32x4*)(Ald + k0 + 8 * hi);      // ds_load_b128
    A.q[1] = *(const u32x4*)(Ald + k0 + 16 + 8 * hi);
#pragma unroll
    for (int t = 0; t < 16; ++t) {
      const int dcol = g * 256 + t * 16 + ln;
      const unsigned short* bp = UbT + (size_t)dcol * QL + k0 + 8 * hi;
      FragBF B;
      B.q[0] = *(const u32x4*)(bp);
      B.q[1] = *(const u32x4*)(bp + 16);
      O[t] = __builtin_amdgcn_wmma_f32_16x16x32_bf16(
          false, A.v, false, B.v, (short)0, O[t], false, false);
    }
  }

  // ---------------- store U_toggler ----------------
#pragma unroll
  for (int t = 0; t < 16; ++t) {
    const int dcol = g * 256 + t * 16 + ln;
#pragma unroll
    for (int r = 0; r < 8; ++r) {
      const size_t row = (size_t)(cbase + rt * 16 + 8 * hi + r);
      outU[row * DD + dcol] = O[t][r];
    }
  }
}

// ---------------------------------------------------------------------------
// reduce_b: softmax stats over bmax[CL]; writes wexp[c] = exp(b-M), stats[0]=1/Z
// single block of 1024 threads
// ---------------------------------------------------------------------------
__global__ void __launch_bounds__(1024) reduceb_kernel(
    const float* __restrict__ bmax, float* __restrict__ wexp,
    float* __restrict__ stats) {
  __shared__ float red[32];
  __shared__ float sM;
  const int t = threadIdx.x;
  float mx = -3.0e38f;
  for (int i = 0; i < CL / 1024; ++i) mx = fmaxf(mx, bmax[t + i * 1024]);
  for (int m = 1; m < 32; m <<= 1) mx = fmaxf(mx, __shfl_xor(mx, m, 32));
  if ((t & 31) == 0) red[t >> 5] = mx;
  __syncthreads();
  if (t < 32) {
    float v = red[t];
    for (int m = 1; m < 32; m <<= 1) v = fmaxf(v, __shfl_xor(v, m, 32));
    if (t == 0) sM = v;
  }
  __syncthreads();
  const float M = sM;
  float s = 0.f;
  for (int i = 0; i < CL / 1024; ++i) {
    const int c = t + i * 1024;
    float e = exp2f((bmax[c] - M) * LOG2E);
    wexp[c] = e;
    s += e;
  }
  for (int m = 1; m < 32; m <<= 1) s += __shfl_xor(s, m, 32);
  __syncthreads();             // red reuse
  if ((t & 31) == 0) red[t >> 5] = s;
  __syncthreads();
  if (t == 0) {
    float z = 0.f;
    for (int i = 0; i < 32; ++i) z += red[i];
    stats[0] = 1.0f / z;
  }
}

// ---------------------------------------------------------------------------
// htog_partial: partial[cc][d] = sum over 256 c of wexp[c]*H[c][d]
// grid: 256 blocks (cc = b>>2 in [0,64), dc = b&3), 256 threads (d)
// ---------------------------------------------------------------------------
__global__ void __launch_bounds__(256) htog_partial_kernel(
    const float* __restrict__ H, const float* __restrict__ wexp,
    float* __restrict__ partial) {
  const int b  = blockIdx.x;
  const int cc = b >> 2;
  const int d  = (b & 3) * 256 + threadIdx.x;
  const int cb = cc * 256;
  float acc = 0.f;
  for (int c = 0; c < 256; ++c)
    acc += wexp[cb + c] * H[(size_t)(cb + c) * DD + d];
  partial[cc * DD + d] = acc;
}

// htog_final: Htog[d] = (1/Z) * sum_cc partial[cc][d]   (grid 4x256)
__global__ void __launch_bounds__(256) htog_final_kernel(
    const float* __restrict__ partial, const float* __restrict__ stats,
    float* __restrict__ Htog) {
  const int d = blockIdx.x * 256 + threadIdx.x;
  float s = 0.f;
  for (int i = 0; i < CL / 256; ++i) s += partial[i * DD + d];
  Htog[d] = s * stats[0];
}

// bcast: outH[c][d] = Htog[d]   (grid 8192 x 256, 8 floats/thread)
__global__ void __launch_bounds__(256) bcast_kernel(
    const float* __restrict__ Htog, float* __restrict__ outH) {
  const size_t base = (size_t)blockIdx.x * 2048 + (size_t)threadIdx.x * 8;
  const int d = (int)(base & (DD - 1));
  f32x4 a = *(const f32x4*)(Htog + d);
  f32x4 b = *(const f32x4*)(Htog + d + 4);
  *(f32x4*)(outH + base)     = a;
  *(f32x4*)(outH + base + 4) = b;
}

// ---------------------------------------------------------------------------
extern "C" void kernel_launch(void* const* d_in, const int* in_sizes, int n_in,
                              void* d_out, int out_size, void* d_ws, size_t ws_size,
                              hipStream_t stream) {
  (void)in_sizes; (void)n_in; (void)out_size; (void)ws_size;
  const float* H    = (const float*)d_in[0];
  const float* U    = (const float*)d_in[1];
  const float* w_q  = (const float*)d_in[2];
  const float* b_q  = (const float*)d_in[3];
  const float* w_c  = (const float*)d_in[4];
  const float* b_c  = (const float*)d_in[5];
  const float* w_qc = (const float*)d_in[6];
  const float* b_qc = (const float*)d_in[7];

  float* outU = (float*)d_out;                       // [CL, DD]
  float* outH = outU + (size_t)CL * DD;              // [CL, DD]

  char* ws = (char*)d_ws;
  unsigned short* Hs   = (unsigned short*)(ws + 0);          // 32 MB
  unsigned short* Ub   = (unsigned short*)(ws + 33554432);   //  2 MB
  unsigned short* UbT  = (unsigned short*)(ws + 35651584);   //  2 MB
  float* uq      = (float*)(ws + 37748736);                  //  4 KB
  float* hc      = (float*)(ws + 37752832);                  // 64 KB
  float* bmax    = (float*)(ws + 37818368);                  // 64 KB
  float* wexp    = (float*)(ws + 37883904);                  // 64 KB
  float* stats   = (float*)(ws + 37949440);                  // 64 B
  float* Htog    = (float*)(ws + 37949504);                  //  4 KB
  float* partial = (float*)(ws + 37953600);                  // 256 KB

  prep_u_kernel<<<QL, 256, 0, stream>>>(U, w_q, b_q, b_c, b_qc, Ub, UbT, uq);
  prep_h_kernel<<<CL, 256, 0, stream>>>(H, w_qc, w_c, Hs, hc);
  attn_kernel<<<CL / 32, 256, 0, stream>>>(Hs, Ub, UbT, uq, hc, outU, bmax);
  reduceb_kernel<<<1, 1024, 0, stream>>>(bmax, wexp, stats);
  htog_partial_kernel<<<256, 256, 0, stream>>>(H, wexp, partial);
  htog_final_kernel<<<4, 256, 0, stream>>>(partial, stats, Htog);
  bcast_kernel<<<CL * DD / 2048, 256, 0, stream>>>(Htog, outH);
}
